// GRUCalibrate_53558242181427
// MI455X (gfx1250) — matrix operand order repaired and checked by reading, not compiled, not run on hardware
//
#include <hip/hip_runtime.h>
#include <hip/hip_bf16.h>
#include <math.h>

// ---------------------------------------------------------------------------
// GNN edge-attention + GRU for MI455X (gfx1250, wave32, WMMA).
// - All GEMMs use v_wmma_f32_16x16x32_f16 (f32 accumulate).
// - Weights are pre-converted once into fragment-native f16 layout so hot
//   loops fetch B-fragments as single 32B vector loads (L2-resident, ~560KB).
// - Edge-level k/v are never materialized: projected tiles are consumed
//   in-register (alpha dot / weighted scatter-add).
// - Node-level projections (~154MB) stay L2-resident (192MB L2) for gathers.
// - Optional f16 edge-fragment cache lets the 2nd edge pass skip f32 cvt.
// ---------------------------------------------------------------------------

typedef __attribute__((ext_vector_type(16))) _Float16 v16h;
typedef __attribute__((ext_vector_type(8)))  float    v8f;

#define DHID  128
#define AHEAD 256
#define G3D   384   // 3 * DHID (GRU gate width)

static __device__ __forceinline__ v8f wmma_f16(v16h a, v16h b, v8f c) {
  return __builtin_amdgcn_wmma_f32_16x16x32_f16(false, a, false, b, (short)0, c,
                                                false, false);
}

// A fragment: 16x32 f16 from f32 row-major. Lanes 0-15 row=lane,
// elems 0-7 = K k0..+7, elems 8-15 = K k0+16..+23; lanes 16-31 K +8/+24.
static __device__ __forceinline__ v16h load_a_frag(const float* __restrict__ A,
                                                   int lda, int row, int k0) {
  const int half = threadIdx.x >> 4;
  const float* p = A + (size_t)row * lda + k0 + half * 8;
  v16h f;
#pragma unroll
  for (int i = 0; i < 8; ++i) f[i] = (_Float16)p[i];
#pragma unroll
  for (int i = 0; i < 8; ++i) f[8 + i] = (_Float16)p[16 + i];
  return f;
}

// Fragment-native weight fetch: Wf[((nt*KstepsTot + ks)*32 + lane)]
static __device__ __forceinline__ v16h load_wf(const v16h* __restrict__ Wf,
                                               int nt, int KstepsTot, int ks) {
  return Wf[((size_t)nt * KstepsTot + ks) * 32 + threadIdx.x];
}

// Ordered-uint encoding of float for atomicMax-based segment max.
static __device__ __forceinline__ unsigned enc_f32(float f) {
  unsigned u = __float_as_uint(f);
  return (u & 0x80000000u) ? ~u : (u | 0x80000000u);
}
static __device__ __forceinline__ float dec_f32(unsigned k) {
  unsigned u = (k & 0x80000000u) ? (k & 0x7fffffffu) : ~k;
  return __uint_as_float(u);
}

// ---------------------------------------------------------------------------
__global__ void init_kernel(unsigned* __restrict__ m_enc,
                            float* __restrict__ denom,
                            float* __restrict__ aggr, int n2, int naggr) {
  int i = blockIdx.x * blockDim.x + threadIdx.x;
  int stride = gridDim.x * blockDim.x;
  for (int j = i; j < n2; j += stride) { m_enc[j] = 0u; denom[j] = 0.0f; }
  for (int j = i; j < naggr; j += stride) aggr[j] = 0.0f;
}

// Convert f32 weight W[K,Nc] into B-fragment-native f16 layout.
// grid = (Nc/16, K/32), 32 threads.
__global__ void __launch_bounds__(32)
weight_frag_kernel(const float* __restrict__ W, v16h* __restrict__ Wf, int Nc) {
  const int lane = threadIdx.x;
  const int nt = blockIdx.x, ks = blockIdx.y;
  const int Ksteps = gridDim.y;
  const int col = nt * 16 + (lane & 15);
  const int kb = ks * 32 + (lane >> 4) * 8;
  const float* p = W + (size_t)kb * Nc + col;
  v16h f;
#pragma unroll
  for (int i = 0; i < 16; ++i) f[i] = (_Float16)p[(size_t)(i < 8 ? i : i + 8) * Nc];
  Wf[((size_t)nt * Ksteps + ks) * 32 + lane] = f;
}

// C[M,Nc] = A[M,K] @ W[K,Nc] (+bias), weights pre-fragmented.
// ksBase selects a K-row window inside Wf (e.g. bottom half of Wk).
__global__ void __launch_bounds__(32)
gemm_wf_kernel(const float* __restrict__ A, const v16h* __restrict__ Wf,
               int KstepsTot, int ksBase, const float* __restrict__ bias,
               float* __restrict__ C, int M, int K, int Nc) {
  const int lane = threadIdx.x;
  const int half = lane >> 4;
  const int coll = lane & 15;
  const int r0 = blockIdx.x * 16;
  const int nt = blockIdx.y;
  int arow = r0 + coll; if (arow > M - 1) arow = M - 1;

  const int Ks = K >> 5;
  v8f acc = {};
  for (int ks = 0; ks < Ks; ++ks) {
    v16h a = load_a_frag(A, K, arow, ks * 32);
    v16h b = load_wf(Wf, nt, KstepsTot, ksBase + ks);
    acc = wmma_f16(a, b, acc);
  }
  const float bvv = bias ? bias[nt * 16 + coll] : 0.0f;
  const int rowbase = r0 + half * 8;
  float* Cp = C + (size_t)rowbase * Nc + nt * 16 + coll;
  if (rowbase + 8 <= M) {       // full tile: straight-line stores
#pragma unroll
    for (int r = 0; r < 8; ++r) Cp[r * Nc] = acc[r] + bvv;
  } else {
#pragma unroll
    for (int r = 0; r < 8; ++r)
      if (rowbase + r < M) Cp[r * Nc] = acc[r] + bvv;
  }
}

// Per 16-edge tile: k = edge_attr@Wk_top + k_node[src] + bk, fused dot with
// gathered q_node[dst] -> alpha[E,2] (scaled+leaky-relu), atomicMax seg max.
// Optionally spills its converted A-fragments to an f16 edge cache.
__global__ void __launch_bounds__(32)
edge_alpha_kernel(const float* __restrict__ edge_attr,
                  const v16h* __restrict__ Wkf, const float* __restrict__ bk,
                  const float* __restrict__ q_node,
                  const float* __restrict__ k_node,
                  const int* __restrict__ src, const int* __restrict__ dst,
                  float* __restrict__ alpha, unsigned* __restrict__ m_enc,
                  v16h* __restrict__ ecache, int E) {
  const int lane = threadIdx.x;
  const int half = lane >> 4;
  const int coll = lane & 15;
  const int e0 = blockIdx.x * 16;
  int arow = e0 + coll; if (arow > E - 1) arow = E - 1;

  if (e0 + 16 < E) {   // stream next edge tile (global_prefetch_b8)
    const float* nf = edge_attr + (size_t)(e0 + 16) * DHID;
#pragma unroll
    for (int i = 0; i < 4; ++i)
      __builtin_prefetch(nf + ((size_t)lane + 32 * i) * 16, 0, 1);
  }

  v16h afr[4];
#pragma unroll
  for (int ks = 0; ks < 4; ++ks)
    afr[ks] = load_a_frag(edge_attr, DHID, arow, ks * 32);
  if (ecache) {
    v16h* ep = ecache + ((size_t)blockIdx.x * 32 + lane) * 4;
#pragma unroll
    for (int ks = 0; ks < 4; ++ks) ep[ks] = afr[ks];
  }

  const int er0 = e0 + half * 8;
  const float* qp[8]; const float* kp[8];
  int dsts[8];
#pragma unroll
  for (int r = 0; r < 8; ++r) {
    int e = er0 + r; if (e > E - 1) e = E - 1;
    dsts[r] = dst[e];
    qp[r] = q_node + (size_t)dsts[r] * AHEAD;
    kp[r] = k_node + (size_t)src[e] * AHEAD;
  }

  float p0[8], p1[8];
#pragma unroll
  for (int r = 0; r < 8; ++r) { p0[r] = 0.0f; p1[r] = 0.0f; }

  for (int nt = 0; nt < 16; ++nt) {
    const int c = nt * 16 + coll;
    v8f acc = {};
#pragma unroll
    for (int ks = 0; ks < 4; ++ks)
      acc = wmma_f16(afr[ks], load_wf(Wkf, nt, 8, ks), acc);
    const float bkc = bk[c];
#pragma unroll
    for (int r = 0; r < 8; ++r) {
      float kv = acc[r] + kp[r][c] + bkc;
      float qv = qp[r][c];
      if (nt < 8) p0[r] += kv * qv; else p1[r] += kv * qv;  // head 0 / head 1
    }
  }
#pragma unroll
  for (int mask = 1; mask < 16; mask <<= 1) {
#pragma unroll
    for (int r = 0; r < 8; ++r) {
      p0[r] += __shfl_xor(p0[r], mask, 32);
      p1[r] += __shfl_xor(p1[r], mask, 32);
    }
  }
  if (coll == 0) {
    const float scale = 0.08838834764831845f;  // 1/sqrt(128)
#pragma unroll
    for (int r = 0; r < 8; ++r) {
      int e = er0 + r;
      if (e < E) {
        float a0 = p0[r] * scale; a0 = (a0 > 0.0f) ? a0 : 0.2f * a0;
        float a1 = p1[r] * scale; a1 = (a1 > 0.0f) ? a1 : 0.2f * a1;
        alpha[(size_t)e * 2 + 0] = a0;
        alpha[(size_t)e * 2 + 1] = a1;
        atomicMax(&m_enc[(size_t)dsts[r] * 2 + 0], enc_f32(a0));
        atomicMax(&m_enc[(size_t)dsts[r] * 2 + 1], enc_f32(a1));
      }
    }
  }
}

// e = exp(alpha - m[dst]); segment-sum denominators.
__global__ void edge_softmax_kernel(const int* __restrict__ dst,
                                    float* __restrict__ alpha,
                                    const unsigned* __restrict__ m_enc,
                                    float* __restrict__ denom, int E) {
  int i = blockIdx.x * blockDim.x + threadIdx.x;
  if (i >= E) return;
  int d = dst[i];
  float m0 = dec_f32(m_enc[(size_t)d * 2 + 0]);
  float m1 = dec_f32(m_enc[(size_t)d * 2 + 1]);
  float e0 = __expf(alpha[(size_t)i * 2 + 0] - m0);
  float e1 = __expf(alpha[(size_t)i * 2 + 1] - m1);
  alpha[(size_t)i * 2 + 0] = e0;
  alpha[(size_t)i * 2 + 1] = e1;
  atomicAdd(&denom[(size_t)d * 2 + 0], e0);
  atomicAdd(&denom[(size_t)d * 2 + 1], e1);
}

// Per 16-edge tile: recompute v = edge_attr@Wv_top + v_node[src] + bv,
// weight by softmax coefficient, scatter-add into aggr[dst, 256].
__global__ void __launch_bounds__(32)
edge_aggr_kernel(const float* __restrict__ edge_attr,
                 const v16h* __restrict__ Wvf, const float* __restrict__ bvp,
                 const float* __restrict__ v_node,
                 const float* __restrict__ eval, const float* __restrict__ denom,
                 const int* __restrict__ src, const int* __restrict__ dst,
                 float* __restrict__ aggr, const v16h* __restrict__ ecache,
                 int E) {
  const int lane = threadIdx.x;
  const int half = lane >> 4;
  const int coll = lane & 15;
  const int e0 = blockIdx.x * 16;
  int arow = e0 + coll; if (arow > E - 1) arow = E - 1;

  v16h afr[4];
  if (ecache) {
    const v16h* ep = ecache + ((size_t)blockIdx.x * 32 + lane) * 4;
    if (e0 + 16 < E)
      __builtin_prefetch(ep + 128, 0, 1);   // next tile of the f16 cache
#pragma unroll
    for (int ks = 0; ks < 4; ++ks) afr[ks] = ep[ks];
  } else {
    if (e0 + 16 < E) {
      const float* nf = edge_attr + (size_t)(e0 + 16) * DHID;
#pragma unroll
      for (int i = 0; i < 4; ++i)
        __builtin_prefetch(nf + ((size_t)lane + 32 * i) * 16, 0, 1);
    }
#pragma unroll
    for (int ks = 0; ks < 4; ++ks)
      afr[ks] = load_a_frag(edge_attr, DHID, arow, ks * 32);
  }

  const int er0 = e0 + half * 8;
  const float* vp[8];
  float* ap[8];
  float a0[8], a1[8];
#pragma unroll
  for (int r = 0; r < 8; ++r) {
    int e = er0 + r; bool valid = (e < E); if (!valid) e = E - 1;
    int d = dst[e];
    vp[r] = v_node + (size_t)src[e] * AHEAD;
    ap[r] = aggr + (size_t)d * AHEAD;
    float w0 = eval[(size_t)e * 2 + 0] / (denom[(size_t)d * 2 + 0] + 1e-16f);
    float w1 = eval[(size_t)e * 2 + 1] / (denom[(size_t)d * 2 + 1] + 1e-16f);
    a0[r] = valid ? w0 : 0.0f;
    a1[r] = valid ? w1 : 0.0f;
  }

  for (int nt = 0; nt < 16; ++nt) {
    const int c = nt * 16 + coll;
    v8f acc = {};
#pragma unroll
    for (int ks = 0; ks < 4; ++ks)
      acc = wmma_f16(afr[ks], load_wf(Wvf, nt, 8, ks), acc);
    const float bvc = bvp[c];
#pragma unroll
    for (int r = 0; r < 8; ++r) {
      float vv = acc[r] + vp[r][c] + bvc;
      float w = (nt < 8) ? a0[r] : a1[r];
      atomicAdd(ap[r] + c, vv * w);
    }
  }
}

// Fused GRU: gates for a 16-node x 16-col tile. h = msg (attention message).
__global__ void __launch_bounds__(32)
gru_kernel(const float* __restrict__ x0, const float* __restrict__ msg,
           const v16h* __restrict__ Wihf, const float* __restrict__ b_ih,
           const v16h* __restrict__ Whhf, const float* __restrict__ b_hh,
           float* __restrict__ out, int Nn) {
  const int lane = threadIdx.x;
  const int half = lane >> 4;
  const int coll = lane & 15;
  const int r0 = blockIdx.x * 16;
  const int ct = blockIdx.y;            // 0..7, column tile within D
  const int c = ct * 16 + coll;
  int arow = r0 + coll; if (arow > Nn - 1) arow = Nn - 1;

  v16h ax[4], am[4];
#pragma unroll
  for (int ks = 0; ks < 4; ++ks) {
    ax[ks] = load_a_frag(x0,  DHID, arow, ks * 32);
    am[ks] = load_a_frag(msg, DHID, arow, ks * 32);
  }
  v8f gir = {}, giz = {}, gin = {}, ghr = {}, ghz = {}, ghn = {};
#pragma unroll
  for (int ks = 0; ks < 4; ++ks) {
    gir = wmma_f16(ax[ks], load_wf(Wihf, ct,      4, ks), gir);
    giz = wmma_f16(ax[ks], load_wf(Wihf, ct + 8,  4, ks), giz);
    gin = wmma_f16(ax[ks], load_wf(Wihf, ct + 16, 4, ks), gin);
    ghr = wmma_f16(am[ks], load_wf(Whhf, ct,      4, ks), ghr);
    ghz = wmma_f16(am[ks], load_wf(Whhf, ct + 8,  4, ks), ghz);
    ghn = wmma_f16(am[ks], load_wf(Whhf, ct + 16, 4, ks), ghn);
  }
  const float bir = b_ih[c], biz = b_ih[DHID + c], bin = b_ih[2 * DHID + c];
  const float bhr = b_hh[c], bhz = b_hh[DHID + c], bhn = b_hh[2 * DHID + c];
  const int rowbase = r0 + half * 8;
  const float* hp = msg + (size_t)rowbase * DHID + c;
  float* op = out + (size_t)rowbase * DHID + c;
  if (rowbase + 8 <= Nn) {
#pragma unroll
    for (int r = 0; r < 8; ++r) {
      float h  = hp[r * DHID];
      float rr = 1.0f / (1.0f + __expf(-(gir[r] + bir + ghr[r] + bhr)));
      float zz = 1.0f / (1.0f + __expf(-(giz[r] + biz + ghz[r] + bhz)));
      float nn = tanhf(gin[r] + bin + rr * (ghn[r] + bhn));
      op[r * DHID] = (1.0f - zz) * nn + zz * h;
    }
  } else {
#pragma unroll
    for (int r = 0; r < 8; ++r) {
      if (rowbase + r < Nn) {
        float h  = hp[r * DHID];
        float rr = 1.0f / (1.0f + __expf(-(gir[r] + bir + ghr[r] + bhr)));
        float zz = 1.0f / (1.0f + __expf(-(giz[r] + biz + ghz[r] + bhz)));
        float nn = tanhf(gin[r] + bin + rr * (ghn[r] + bhn));
        op[r * DHID] = (1.0f - zz) * nn + zz * h;
      }
    }
  }
}

// ---------------------------------------------------------------------------
extern "C" void kernel_launch(void* const* d_in, const int* in_sizes, int n_in,
                              void* d_out, int out_size, void* d_ws,
                              size_t ws_size, hipStream_t stream) {
  const float* x0        = (const float*)d_in[0];
  const float* x1        = (const float*)d_in[1];
  const float* edge_attr = (const float*)d_in[2];
  const int*   edges     = (const int*)d_in[3];
  /* d_in[4] = edge_types, unused by the reference computation */
  const float* Wq   = (const float*)d_in[5];
  const float* bq   = (const float*)d_in[6];
  const float* Wk   = (const float*)d_in[7];
  const float* bk   = (const float*)d_in[8];
  const float* Wv   = (const float*)d_in[9];
  const float* bv   = (const float*)d_in[10];
  const float* Wo   = (const float*)d_in[11];
  const float* bo   = (const float*)d_in[12];
  const float* W_ih = (const float*)d_in[13];
  const float* b_ih = (const float*)d_in[14];
  const float* W_hh = (const float*)d_in[15];
  const float* b_hh = (const float*)d_in[16];

  const int N = in_sizes[0] / DHID;
  const int E = in_sizes[4];
  const int* src = edges;        // edges[0]
  const int* dst = edges + E;    // edges[1]

  // ---- workspace layout: f32 region, then f16 region (32B-aligned chunks)
  float* ws = (float*)d_ws;
  size_t off = 0;
  float* q_node = ws + off; off += (size_t)N * AHEAD;
  float* k_node = ws + off; off += (size_t)N * AHEAD;
  float* v_node = ws + off; off += (size_t)N * AHEAD;
  float* aggr   = ws + off; off += (size_t)N * AHEAD;
  float* msg    = ws + off; off += (size_t)N * DHID;
  float* eval   = ws + off; off += (size_t)E * 2;
  float* denom  = ws + off; off += (size_t)N * 2;
  unsigned* m_enc = (unsigned*)(ws + off); off += (size_t)N * 2;

  _Float16* hbase = (_Float16*)(ws + off);
  size_t offh = 0;
  v16h* Wq_f  = (v16h*)(hbase + offh); offh += (size_t)DHID * AHEAD;
  v16h* Wk_f  = (v16h*)(hbase + offh); offh += (size_t)AHEAD * AHEAD;
  v16h* Wv_f  = (v16h*)(hbase + offh); offh += (size_t)AHEAD * AHEAD;
  v16h* Wo_f  = (v16h*)(hbase + offh); offh += (size_t)AHEAD * DHID;
  v16h* Wih_f = (v16h*)(hbase + offh); offh += (size_t)DHID * G3D;
  v16h* Whh_f = (v16h*)(hbase + offh); offh += (size_t)DHID * G3D;
  v16h* ecache = (v16h*)(hbase + offh);
  const size_t need_base  = off * 4 + offh * 2;
  const size_t need_cache = need_base + (size_t)E * DHID * 2;
  v16h* ec = (ws_size >= need_cache) ? ecache : (v16h*)nullptr;

  const int mt_n = (N + 15) / 16;
  const int mt_e = (E + 15) / 16;

  init_kernel<<<1024, 256, 0, stream>>>(m_enc, denom, aggr, N * 2, N * AHEAD);

  // One-shot weight conversion into fragment-native f16 layout (~560KB).
  weight_frag_kernel<<<dim3(AHEAD / 16, DHID / 32), 32, 0, stream>>>(Wq,  Wq_f,  AHEAD);
  weight_frag_kernel<<<dim3(AHEAD / 16, AHEAD / 32), 32, 0, stream>>>(Wk,  Wk_f,  AHEAD);
  weight_frag_kernel<<<dim3(AHEAD / 16, AHEAD / 32), 32, 0, stream>>>(Wv,  Wv_f,  AHEAD);
  weight_frag_kernel<<<dim3(DHID / 16,  AHEAD / 32), 32, 0, stream>>>(Wo,  Wo_f,  DHID);
  weight_frag_kernel<<<dim3(G3D / 16,   DHID / 32), 32, 0, stream>>>(W_ih, Wih_f, G3D);
  weight_frag_kernel<<<dim3(G3D / 16,   DHID / 32), 32, 0, stream>>>(W_hh, Whh_f, G3D);

  // Node-level projections (hoisted out of the edge loop; L2-resident).
  dim3 g1(mt_n, AHEAD / 16);
  gemm_wf_kernel<<<g1, 32, 0, stream>>>(x0, Wq_f, 4, 0, bq, q_node, N, DHID, AHEAD);
  gemm_wf_kernel<<<g1, 32, 0, stream>>>(x1, Wk_f, 8, 4, nullptr, k_node, N, DHID, AHEAD);
  gemm_wf_kernel<<<g1, 32, 0, stream>>>(x1, Wv_f, 8, 4, nullptr, v_node, N, DHID, AHEAD);

  // Edge attention scores + segment max (spills f16 edge frags if ws allows).
  edge_alpha_kernel<<<mt_e, 32, 0, stream>>>(edge_attr, Wk_f, bk, q_node,
                                             k_node, src, dst, eval, m_enc,
                                             ec, E);
  // Softmax numerators + segment sums.
  edge_softmax_kernel<<<(E + 255) / 256, 256, 0, stream>>>(dst, eval, m_enc,
                                                           denom, E);
  // Weighted value aggregation (v recomputed, never materialized).
  edge_aggr_kernel<<<mt_e, 32, 0, stream>>>(edge_attr, Wv_f, bv, v_node, eval,
                                            denom, src, dst, aggr, ec, E);
  // Output projection.
  dim3 g2(mt_n, DHID / 16);
  gemm_wf_kernel<<<g2, 32, 0, stream>>>(aggr, Wo_f, 16, 0, bo, msg, N, AHEAD, DHID);
  // Fused GRU cell.
  dim3 g3(mt_n, DHID / 16);
  gru_kernel<<<g3, 32, 0, stream>>>(x0, msg, Wih_f, b_ih, Whh_f, b_hh,
                                    (float*)d_out, N);
}
